// Model_on_CGCNN_41420664602958
// MI455X (gfx1250) — compile-verified, matrix operand order checked by reading, not run on hardware
//
#include <hip/hip_runtime.h>

typedef __bf16 bf16;
typedef __attribute__((ext_vector_type(16))) __bf16 v16bf;
typedef __attribute__((ext_vector_type(8)))  __bf16 v8bf;
typedef __attribute__((ext_vector_type(8)))  float  v8f;
typedef __attribute__((ext_vector_type(4)))  unsigned int v4u;
typedef __attribute__((ext_vector_type(8)))  unsigned int v8u;

#define TM 128
#define TN 128
#define TK 32
#define USE_TDM 1

// ---------------------------------------------------------------------------
// Tensor Data Mover: issue a 2D tile load (128 rows x 32 bf16, row stride ld)
// D# per CDNA5 ISA. Descriptor dims == tile dims
// (buffers are fully padded, so no OOB handling needed).
// ---------------------------------------------------------------------------
__device__ __forceinline__ void tdm_load_tile(unsigned lds_off, const bf16* gptr,
                                              unsigned ld /*elements*/)
{
    unsigned long long ga = (unsigned long long)(size_t)gptr;
    v4u g0; v8u g1;
    g0[0] = 1u;                                            // count=1 user D#
    g0[1] = lds_off;                                       // lds_addr (bytes)
    g0[2] = (unsigned)ga;                                  // global_addr[31:0]
    g0[3] = ((unsigned)(ga >> 32) & 0x01FFFFFFu)           // global_addr[56:32]
          | (2u << 30);                                    // type = 2 (image)
    g1[0] = (1u << 16);                                    // data_size=1 -> 2B
    g1[1] = (32u & 0xFFFFu) << 16;                         // tensor_dim0=32 [63:48]
    g1[2] = 0u | ((128u & 0xFFFFu) << 16);                 // td0 hi | tensor_dim1=128
    g1[3] = 0u | (32u << 16);                              // td1 hi | tile_dim0=32
    g1[4] = 128u;                                          // tile_dim1=128, tile_dim2=0
    g1[5] = ld;                                            // tensor_dim0_stride[31:0]
    g1[6] = 0u;                                            // stride0 hi | dim1_stride lo
    g1[7] = 0u;                                            // dim1_stride hi
#pragma unroll
    for (int i = 0; i < 4; ++i) g0[i] = __builtin_amdgcn_readfirstlane(g0[i]);
#pragma unroll
    for (int i = 0; i < 8; ++i) g1[i] = __builtin_amdgcn_readfirstlane(g1[i]);
    asm volatile("tensor_load_to_lds %0, %1" :: "s"(g0), "s"(g1) : "memory");
}

// ---------------------------------------------------------------------------
// Generic bf16 WMMA GEMM: C[M,N] = A[M,K] @ B^T(stored [N,K]) (+bias)(+relu)
// A : bf16 row-major [Mpad, lda], K padded to mult of 32
// Bt: bf16 row-major [Npad, ldb]  (pre-transposed weights)
// C : f32 row-major, guarded stores to (Mreal, Nreal)
// Double-buffered LDS; tiles fetched by the Tensor Data Mover (TENSORcnt).
// ---------------------------------------------------------------------------
__global__ __launch_bounds__(256) void gemm_bf16_wmma(
    const bf16* __restrict__ A, int lda,
    const bf16* __restrict__ Bt, int ldb,
    float* __restrict__ C, int ldc,
    int Mreal, int Nreal, int K,
    const float* __restrict__ bias, int relu)
{
    __shared__ __align__(16) bf16 sA[2][TM * TK];   // [m][k]
    __shared__ __align__(16) bf16 sB[2][TN * TK];   // [n][k]

    const int tid  = threadIdx.x;
    const int wave = tid >> 5;
    const int lane = tid & 31;
    const int wm   = wave & 3;       // 4 waves along M
    const int wn   = wave >> 2;      // 2 waves along N
    const int fr   = lane & 15;
    const int fh   = lane >> 4;

    const long tileM = (long)blockIdx.x * TM;
    const long tileN = (long)blockIdx.y * TN;
    const bf16* Abase = A  + tileM * (long)lda;
    const bf16* Bbase = Bt + tileN * (long)ldb;

    v8f acc[2][4];
#pragma unroll
    for (int i = 0; i < 2; ++i)
#pragma unroll
        for (int j = 0; j < 4; ++j)
#pragma unroll
            for (int e = 0; e < 8; ++e)
                acc[i][j][e] = 0.f;

    const int nk = K / TK;

#if USE_TDM
    const unsigned offA0 = (unsigned)(size_t)&sA[0][0];
    const unsigned offA1 = (unsigned)(size_t)&sA[1][0];
    const unsigned offB0 = (unsigned)(size_t)&sB[0][0];
    const unsigned offB1 = (unsigned)(size_t)&sB[1][0];
    // prologue: DMA tile 0 into buffer 0
    if (wave == 0) {
        tdm_load_tile(offA0, Abase, (unsigned)lda);
        tdm_load_tile(offB0, Bbase, (unsigned)ldb);
        __builtin_amdgcn_s_wait_tensorcnt(0);
    }
    __syncthreads();
#endif

    for (int kt = 0; kt < nk; ++kt) {
        const int cur = kt & 1;
        const int k0  = kt * TK;

#if USE_TDM
        // prefetch next K-tile into the other buffer while we compute
        if (wave == 0 && kt + 1 < nk) {
            int k1 = k0 + TK;
            tdm_load_tile(cur ? offA0 : offA1, Abase + k1, (unsigned)lda);
            tdm_load_tile(cur ? offB0 : offB1, Bbase + k1, (unsigned)ldb);
        }
#else
        // synchronous staging: both tiles are contiguous 128x32 blocks
#pragma unroll
        for (int i = 0; i < 2; ++i) {
            int chunk = tid + i * 256;            // 0..511
            int row   = chunk >> 2;
            int kc    = (chunk & 3) * 8;
            *(uint4*)(&sA[cur][row * TK + kc]) =
                *(const uint4*)(Abase + (long)row * lda + k0 + kc);
            *(uint4*)(&sB[cur][row * TK + kc]) =
                *(const uint4*)(Bbase + (long)row * ldb + k0 + kc);
        }
        __syncthreads();
#endif

        // ---- assemble fragments per ISA 16-bit A/B layout and issue WMMA ----
        // lanes 0-15 hold K{0..7,16..23}, lanes 16-31 hold K{8..15,24..31}
        v16bf afrag[2];
#pragma unroll
        for (int mf = 0; mf < 2; ++mf) {
            const bf16* pa = &sA[cur][(wm * 32 + mf * 16 + fr) * TK];
            v8bf lo = *(const v8bf*)(pa + 8 * fh);
            v8bf hi = *(const v8bf*)(pa + 16 + 8 * fh);
            afrag[mf] = __builtin_shufflevector(lo, hi, 0,1,2,3,4,5,6,7,8,9,10,11,12,13,14,15);
        }
        v16bf bfrag[4];
#pragma unroll
        for (int nf = 0; nf < 4; ++nf) {
            const bf16* pb = &sB[cur][(wn * 64 + nf * 16 + fr) * TK];
            v8bf lo = *(const v8bf*)(pb + 8 * fh);
            v8bf hi = *(const v8bf*)(pb + 16 + 8 * fh);
            bfrag[nf] = __builtin_shufflevector(lo, hi, 0,1,2,3,4,5,6,7,8,9,10,11,12,13,14,15);
        }
#pragma unroll
        for (int mf = 0; mf < 2; ++mf)
#pragma unroll
            for (int nf = 0; nf < 4; ++nf)
                acc[mf][nf] = __builtin_amdgcn_wmma_f32_16x16x32_bf16(
                    false, afrag[mf], false, bfrag[nf],
                    (short)0, acc[mf][nf], false, false);

#if USE_TDM
        if (wave == 0 && kt + 1 < nk)
            __builtin_amdgcn_s_wait_tensorcnt(0);   // next tile landed
#endif
        __syncthreads();
    }

    // ---- epilogue: C/D layout -> row = v + 8*fh within fragment ----
#pragma unroll
    for (int mf = 0; mf < 2; ++mf) {
#pragma unroll
        for (int nf = 0; nf < 4; ++nf) {
            long col = tileN + wn * 64 + nf * 16 + fr;
            if (col >= Nreal) continue;
            float bv = bias ? bias[col] : 0.f;
#pragma unroll
            for (int v = 0; v < 8; ++v) {
                long row = tileM + wm * 32 + mf * 16 + fh * 8 + v;
                if (row < Mreal) {
                    float val = acc[mf][nf][v] + bv;
                    if (relu) val = fmaxf(val, 0.f);
                    C[row * (long)ldc + col] = val;
                }
            }
        }
    }
}

// ---------------------------------------------------------------------------
// Elementwise / reduction helper kernels
// ---------------------------------------------------------------------------
// pack weight TRANSPOSED: out[n*Kpad + k] = W[k*N + n], zero-padded
__global__ void k_pack_weight_T(const float* __restrict__ W, bf16* __restrict__ out,
                                int K, int N, int Kpad, int Npad)
{
    long i = (long)blockIdx.x * blockDim.x + threadIdx.x;
    long total = (long)Kpad * Npad;
    if (i >= total) return;
    int n = (int)(i / Kpad);
    int k = (int)(i % Kpad);
    float v = (k < K && n < N) ? W[(long)k * N + n] : 0.f;
    out[i] = (bf16)v;
}

__global__ void k_pack_act(const float* __restrict__ X, bf16* __restrict__ out,
                           long M, int K, int Kpad)
{
    long i = (long)blockIdx.x * blockDim.x + threadIdx.x;
    long total = M * Kpad;
    if (i >= total) return;
    long r = i / Kpad;
    int  c = (int)(i % Kpad);
    float v = (c < K) ? X[r * K + c] : 0.f;
    out[i] = (bf16)v;
}

__global__ void k_rbf(const float* __restrict__ r3, bf16* __restrict__ out, int E)
{
    int e = blockIdx.x * blockDim.x + threadIdx.x;
    if (e >= E) return;
    float x = r3[3 * e + 0], y = r3[3 * e + 1], z = r3[3 * e + 2];
    float d = sqrtf(x * x + y * y + z * z);
    const float gamma = 79.f / 8.f;
    bf16* o = out + (long)e * 96;
    for (int b = 0; b < 96; ++b) {
        float val = 0.f;
        if (b < 80) {
            float t = d - (8.f / 79.f) * (float)b;
            val = expf(-gamma * t * t);
        }
        o[b] = (bf16)val;
    }
}

__global__ void k_bn_stats(const float* __restrict__ X, float* __restrict__ sum,
                           float* __restrict__ sq, long M, int C, int rows)
{
    long r0 = (long)blockIdx.x * rows;
    long r1 = r0 + rows; if (r1 > M) r1 = M;
    for (int c = threadIdx.x; c < C; c += blockDim.x) {
        float s = 0.f, q = 0.f;
        for (long r = r0; r < r1; ++r) {
            float v = X[r * C + c];
            s += v; q += v * v;
        }
        atomicAdd(&sum[c], s);
        atomicAdd(&sq[c], q);
    }
}

__global__ void k_bn_finalize(const float* __restrict__ sum, const float* __restrict__ sq,
                              const float* __restrict__ g, const float* __restrict__ b,
                              float* __restrict__ scale, float* __restrict__ shift,
                              float invM, int C)
{
    int c = blockIdx.x * blockDim.x + threadIdx.x;
    if (c >= C) return;
    float mu  = sum[c] * invM;
    float var = sq[c] * invM - mu * mu;
    float s   = g[c] * rsqrtf(var + 1e-5f);
    scale[c] = s;
    shift[c] = b[c] - mu * s;
}

__global__ void k_bn_relu_pack(const float* __restrict__ Z, const float* __restrict__ scale,
                               const float* __restrict__ shift,
                               float* __restrict__ outF, bf16* __restrict__ outB,
                               long total, int C)
{
    long i = (long)blockIdx.x * blockDim.x + threadIdx.x;
    if (i >= total) return;
    int c = (int)(i % C);
    float v = fmaxf(fmaf(Z[i], scale[c], shift[c]), 0.f);
    if (outF) outF[i] = v;
    outB[i] = (bf16)v;
}

// m[e,:] += hs[src[e],:] + hd[dst[e],:], accumulating per-channel BN stats.
__global__ __launch_bounds__(256) void k_edge_combine(
    float* __restrict__ m, const float* __restrict__ hs, const float* __restrict__ hd,
    const int* __restrict__ src, const int* __restrict__ dst,
    float* __restrict__ sum, float* __restrict__ sq, int E)
{
    int t = threadIdx.x;
    long e0 = (long)blockIdx.x * 32;
    int c0 = t, c1 = t + 256;
    float s0 = 0.f, q0 = 0.f, s1 = 0.f, q1 = 0.f;
    for (int r = 0; r < 32; ++r) {
        long e = e0 + r;
        if (e >= E) break;
        int se = src[e], de = dst[e];
        float v0 = m[e * 512 + c0] + hs[(long)se * 512 + c0] + hd[(long)de * 512 + c0];
        float v1 = m[e * 512 + c1] + hs[(long)se * 512 + c1] + hd[(long)de * 512 + c1];
        m[e * 512 + c0] = v0;
        m[e * 512 + c1] = v1;
        s0 += v0; q0 += v0 * v0;
        s1 += v1; q1 += v1 * v1;
    }
    atomicAdd(&sum[c0], s0); atomicAdd(&sq[c0], q0);
    atomicAdd(&sum[c1], s1); atomicAdd(&sq[c1], q1);
}

// gated message + segment-sum scatter to destination nodes (f32 atomics)
__global__ void k_gate_scatter(const float* __restrict__ m, const float* __restrict__ scale,
                               const float* __restrict__ shift, const int* __restrict__ dst,
                               float* __restrict__ hnode, long total)
{
    long i = (long)blockIdx.x * blockDim.x + threadIdx.x;
    if (i >= total) return;
    long e = i >> 8;
    int  c = (int)(i & 255);
    float hf  = fmaf(m[e * 512 + c],       scale[c],       shift[c]);
    float hsp = fmaf(m[e * 512 + 256 + c], scale[256 + c], shift[256 + c]);
    float sg = 1.f / (1.f + expf(-hf));
    float sp = (hsp > 20.f) ? hsp : log1pf(expf(hsp));
    atomicAdd(&hnode[(long)dst[e] * 256 + c], sg * sp);
}

__global__ void k_node_update(float* __restrict__ x, const float* __restrict__ h,
                              const float* __restrict__ scale, const float* __restrict__ shift,
                              bf16* __restrict__ xb, long total, int C)
{
    long i = (long)blockIdx.x * blockDim.x + threadIdx.x;
    if (i >= total) return;
    int c = (int)(i % C);
    float v = x[i] + fmaf(h[i], scale[c], shift[c]);
    v = (v > 20.f) ? v : log1pf(expf(v));
    x[i] = v;
    xb[i] = (bf16)v;
}

// ---------------------------------------------------------------------------
// Host orchestration
// ---------------------------------------------------------------------------
static inline unsigned gblk(long total, int bs) { return (unsigned)((total + bs - 1) / bs); }

extern "C" void kernel_launch(void* const* d_in, const int* in_sizes, int n_in,
                              void* d_out, int out_size, void* d_ws, size_t ws_size,
                              hipStream_t stream)
{
    const float* atomf = (const float*)d_in[0];
    const float* rpos  = (const float*)d_in[1];
    const int*   src   = (const int*)d_in[2];
    const int*   dst   = (const int*)d_in[3];
    const float* aW    = (const float*)d_in[4];
    const float* ag    = (const float*)d_in[6];
    const float* abeta = (const float*)d_in[7];
    const float* e1W   = (const float*)d_in[8];
    const float* e1g   = (const float*)d_in[10];
    const float* e1beta= (const float*)d_in[11];
    const float* e2W   = (const float*)d_in[12];
    const float* e2g   = (const float*)d_in[14];
    const float* e2beta= (const float*)d_in[15];
    const float* cWs   = (const float*)d_in[16];
    const float* cWd   = (const float*)d_in[18];
    const float* cWe   = (const float*)d_in[20];
    const float* cmg   = (const float*)d_in[22];
    const float* cmb   = (const float*)d_in[23];
    const float* cng   = (const float*)d_in[24];
    const float* cnb   = (const float*)d_in[25];
    const float* f1W   = (const float*)d_in[26];
    const float* f1b   = (const float*)d_in[27];
    const float* f2W   = (const float*)d_in[28];
    const float* f2b   = (const float*)d_in[29];
    const float* f3W   = (const float*)d_in[30];
    const float* f3b   = (const float*)d_in[31];
    float* out = (float*)d_out;

    const int FA = 92, FAp = 96, BINSp = 96, EMB = 64, H = 256, H2 = 512, L = 4;
    const long Nn = in_sizes[0] / FA;             // 10000
    const int  E  = in_sizes[2];                  // 160000
    const long NPAD = ((Nn + 127) / 128) * 128;   // 10112
    const int  FN = 10000, FNP = 10112;

    size_t off = 0;
    auto alloc = [&](size_t n) -> char* {
        char* p = (char*)d_ws + off;
        off = (off + n + 255) & ~(size_t)255;
        return p;
    };
    // weights stored TRANSPOSED: [Npad rows][Kpad cols]
    bf16* aWb  = (bf16*)alloc((size_t)H * FAp * 2);
    bf16* e1Wb = (bf16*)alloc((size_t)EMB * BINSp * 2);
    bf16* e2Wb = (bf16*)alloc((size_t)H * EMB * 2);
    bf16* cWsb = (bf16*)alloc((size_t)L * H2 * H * 2);
    bf16* cWdb = (bf16*)alloc((size_t)L * H2 * H * 2);
    bf16* cWeb = (bf16*)alloc((size_t)L * H2 * H * 2);
    bf16* f1Wb = (bf16*)alloc((size_t)1024 * H * 2);
    bf16* f2Wb = (bf16*)alloc((size_t)2048 * 1024 * 2);
    bf16* f3Wb = (bf16*)alloc((size_t)FNP * 2048 * 2);
    bf16* atomb= (bf16*)alloc((size_t)NPAD * FAp * 2);
    bf16* rbfb = (bf16*)alloc((size_t)E * BINSp * 2);
    bf16* y1b  = (bf16*)alloc((size_t)E * EMB * 2);
    bf16* yb   = (bf16*)alloc((size_t)E * H * 2);
    bf16* xb   = (bf16*)alloc((size_t)NPAD * H * 2);
    float* xf  = (float*)alloc((size_t)Nn * H * 4);
    float* hsB = (float*)alloc((size_t)Nn * H2 * 4);
    float* hdB = (float*)alloc((size_t)Nn * H2 * 4);
    float* hnode = (float*)alloc((size_t)Nn * H * 4);
    float* ssum  = (float*)alloc(2 * 512 * 4);
    float* ssq   = ssum + 512;
    float* sscale= (float*)alloc(2 * 512 * 4);
    float* sshift= sscale + 512;
    char* BIG = alloc((size_t)E * H2 * 4);          // 328 MB, multi-purpose
    float* zA   = (float*)BIG;
    float* zE1  = (float*)BIG;
    float* zE2  = (float*)BIG;
    float* mbuf = (float*)BIG;
    float* z1   = (float*)BIG;
    bf16*  z1b  = (bf16*)(BIG + ((size_t)64 << 20));
    float* z2   = (float*)(BIG + ((size_t)96 << 20));
    bf16*  z2b  = (bf16*)(BIG + ((size_t)192 << 20));

    auto gemm = [&](const bf16* A, int lda, const bf16* Bt, int ldb, float* Cm, int ldc,
                    long M, long Nc, int K, const float* bias, int relu) {
        dim3 grid((unsigned)((M + TM - 1) / TM), (unsigned)((Nc + TN - 1) / TN));
        gemm_bf16_wmma<<<grid, 256, 0, stream>>>(A, lda, Bt, ldb, Cm, ldc,
                                                 (int)M, (int)Nc, K, bias, relu);
    };
    auto zero_stats = [&]() { hipMemsetAsync(ssum, 0, 2 * 512 * 4, stream); };

    // ---- pack weights to bf16, transposed + zero-padded ----
    k_pack_weight_T<<<gblk((long)FAp * H, 256), 256, 0, stream>>>(aW, aWb, FA, H, FAp, H);
    k_pack_weight_T<<<gblk((long)BINSp * EMB, 256), 256, 0, stream>>>(e1W, e1Wb, 80, EMB, BINSp, EMB);
    k_pack_weight_T<<<gblk((long)EMB * H, 256), 256, 0, stream>>>(e2W, e2Wb, EMB, H, EMB, H);
    for (int l = 0; l < L; ++l) {
        size_t o = (size_t)l * H * H2;
        k_pack_weight_T<<<gblk((long)H * H2, 256), 256, 0, stream>>>(cWs + o, cWsb + o, H, H2, H, H2);
        k_pack_weight_T<<<gblk((long)H * H2, 256), 256, 0, stream>>>(cWd + o, cWdb + o, H, H2, H, H2);
        k_pack_weight_T<<<gblk((long)H * H2, 256), 256, 0, stream>>>(cWe + o, cWeb + o, H, H2, H, H2);
    }
    k_pack_weight_T<<<gblk((long)H * 1024, 256), 256, 0, stream>>>(f1W, f1Wb, H, 1024, H, 1024);
    k_pack_weight_T<<<gblk((long)1024 * 2048, 256), 256, 0, stream>>>(f2W, f2Wb, 1024, 2048, 1024, 2048);
    k_pack_weight_T<<<gblk((long)2048 * FNP, 256), 256, 0, stream>>>(f3W, f3Wb, 2048, FN, 2048, FNP);

    // ---- atom embedding: x = relu(BN(atom @ aW)) ----
    k_pack_act<<<gblk(Nn * FAp, 256), 256, 0, stream>>>(atomf, atomb, Nn, FA, FAp);
    gemm(atomb, FAp, aWb, FAp, zA, H, Nn, H, FAp, nullptr, 0);
    zero_stats();
    k_bn_stats<<<gblk(Nn, 128), 256, 0, stream>>>(zA, ssum, ssq, Nn, H, 128);
    k_bn_finalize<<<1, 256, 0, stream>>>(ssum, ssq, ag, abeta, sscale, sshift, 1.f / Nn, H);
    k_bn_relu_pack<<<gblk(Nn * H, 256), 256, 0, stream>>>(zA, sscale, sshift, xf, xb, Nn * H, H);

    // ---- edge embedding: rbf -> MLP(80->64) -> MLP(64->256) ----
    k_rbf<<<gblk(E, 256), 256, 0, stream>>>(rpos, rbfb, E);
    gemm(rbfb, BINSp, e1Wb, BINSp, zE1, EMB, E, EMB, BINSp, nullptr, 0);
    zero_stats();
    k_bn_stats<<<gblk(E, 128), 256, 0, stream>>>(zE1, ssum, ssq, E, EMB, 128);
    k_bn_finalize<<<1, 256, 0, stream>>>(ssum, ssq, e1g, e1beta, sscale, sshift, 1.f / E, EMB);
    k_bn_relu_pack<<<gblk((long)E * EMB, 256), 256, 0, stream>>>(zE1, sscale, sshift, nullptr, y1b, (long)E * EMB, EMB);
    gemm(y1b, EMB, e2Wb, EMB, zE2, H, E, H, EMB, nullptr, 0);
    zero_stats();
    k_bn_stats<<<gblk(E, 128), 256, 0, stream>>>(zE2, ssum, ssq, E, H, 128);
    k_bn_finalize<<<1, 256, 0, stream>>>(ssum, ssq, e2g, e2beta, sscale, sshift, 1.f / E, H);
    k_bn_relu_pack<<<gblk((long)E * H, 256), 256, 0, stream>>>(zE2, sscale, sshift, nullptr, yb, (long)E * H, H);

    // ---- CGCNN conv layers ----
    for (int l = 0; l < L; ++l) {
        size_t o = (size_t)l * H * H2;
        gemm(xb, H, cWsb + o, H, hsB, H2, Nn, H2, H, nullptr, 0);
        gemm(xb, H, cWdb + o, H, hdB, H2, Nn, H2, H, nullptr, 0);
        gemm(yb, H, cWeb + o, H, mbuf, H2, E, H2, H, nullptr, 0);
        zero_stats();
        k_edge_combine<<<gblk(E, 32), 256, 0, stream>>>(mbuf, hsB, hdB, src, dst, ssum, ssq, E);
        k_bn_finalize<<<2, 256, 0, stream>>>(ssum, ssq, cmg + (size_t)l * H2, cmb + (size_t)l * H2,
                                             sscale, sshift, 1.f / E, H2);
        hipMemsetAsync(hnode, 0, (size_t)Nn * H * 4, stream);
        k_gate_scatter<<<gblk((long)E * H, 256), 256, 0, stream>>>(mbuf, sscale, sshift, dst, hnode, (long)E * H);
        zero_stats();
        k_bn_stats<<<gblk(Nn, 128), 256, 0, stream>>>(hnode, ssum, ssq, Nn, H, 128);
        k_bn_finalize<<<1, 256, 0, stream>>>(ssum, ssq, cng + (size_t)l * H, cnb + (size_t)l * H,
                                             sscale, sshift, 1.f / Nn, H);
        k_node_update<<<gblk(Nn * H, 256), 256, 0, stream>>>(xf, hnode, sscale, sshift, xb, Nn * H, H);
    }

    // ---- dense head ----
    gemm(xb, H, f1Wb, H, z1, 1024, Nn, 1024, H, f1b, 1);
    k_pack_act<<<gblk(Nn * 1024, 256), 256, 0, stream>>>(z1, z1b, Nn, 1024, 1024);
    gemm(z1b, 1024, f2Wb, 1024, z2, 2048, Nn, 2048, 1024, f2b, 1);
    k_pack_act<<<gblk(Nn * 2048, 256), 256, 0, stream>>>(z2, z2b, Nn, 2048, 2048);
    gemm(z2b, 2048, f3Wb, 2048, out, FN, Nn, FN, 2048, f3b, 1);  // relu(relu(x)) == relu(x)
}
